// Multi_EfficientAttention_4321_56229711839615
// MI455X (gfx1250) — matrix operand order repaired and checked by reading, not compiled, not run on hardware
//
#include <hip/hip_runtime.h>
#include <hip/hip_bf16.h>

// Problem dims (fixed by reference)
#define BSZ   4
#define SEQ   8192
#define DIM   256
#define HEADS 8
#define HD    32
#define MROWS (BSZ * SEQ)   // 32768

typedef __attribute__((ext_vector_type(16))) __bf16 bf16x16;
typedef __attribute__((ext_vector_type(8)))  __bf16 bf16x8;
typedef __attribute__((ext_vector_type(8)))  float  f32x8;
typedef __attribute__((ext_vector_type(4)))  float  f32x4;
typedef __attribute__((ext_vector_type(4)))  unsigned short u16x4;

static __device__ __forceinline__ unsigned short f2bf(float f) {
  unsigned int u = __builtin_bit_cast(unsigned int, f);
  u += 0x7fffu + ((u >> 16) & 1u);           // round-to-nearest-even
  return (unsigned short)(u >> 16);
}

// A-fragment (16x32 bf16, M x K), row-major source with element stride ld.
// Contiguous k order, same as B: lanes 0-15 hold M=lane, K 0..15; lanes 16-31 K 16..31.
static __device__ __forceinline__ bf16x16 load_a_frag(const unsigned short* base, int ld) {
  const int lane = threadIdx.x & 31;
  const int m = lane & 15, hf = lane >> 4;
  const unsigned short* p = base + (size_t)m * ld + hf * 16;
  bf16x8 lo = *(const bf16x8*)(p);
  bf16x8 hi = *(const bf16x8*)(p + 8);
  return __builtin_shufflevector(lo, hi, 0,1,2,3,4,5,6,7,8,9,10,11,12,13,14,15);
}

// B-fragment (32x16 bf16, K x N) supplied TRANSPOSED (N rows of K, stride ld).
// Contiguous k order, same as A: lanes 0-15 hold N=lane, K 0..15; lanes 16-31 K 16..31. One 32B load.
static __device__ __forceinline__ bf16x16 load_b_frag(const unsigned short* baseT, int ld) {
  const int lane = threadIdx.x & 31;
  const int n = lane & 15, hf = lane >> 4;
  return *(const bf16x16*)(baseT + (size_t)n * ld + hf * 16);
}

static __device__ __forceinline__ f32x8 wmma_bf16(bf16x16 a, bf16x16 b, f32x8 c) {
  return __builtin_amdgcn_wmma_f32_16x16x32_bf16(false, a, false, b, (short)0, c, false, false);
}

// ---------------------------------------------------------------- K0: weights
__global__ __launch_bounds__(256) void prep_weights(
    const float* __restrict__ Wq, const float* __restrict__ Wk,
    const float* __restrict__ Wv, const float* __restrict__ Wrp,
    unsigned short* __restrict__ wqkvT, unsigned short* __restrict__ wrpT) {
  int idx = blockIdx.x * 256 + threadIdx.x;            // < 786432
  const int QKV = 9 * DIM * DIM;
  if (idx < QKV) {
    int g = idx >> 16;
    int r = idx & 65535;
    int k = r >> 8, n = r & 255;
    const float* src = (g < 3) ? (Wq + (size_t)g * DIM * DIM)
                     : (g < 6) ? (Wk + (size_t)(g - 3) * DIM * DIM)
                               : (Wv + (size_t)(g - 6) * DIM * DIM);
    wqkvT[(size_t)g * DIM * DIM + (size_t)n * DIM + k] = f2bf(src[k * DIM + n]);
  } else {
    int r = idx - QKV;                                  // < 768*256
    int k = r >> 8, n = r & 255;
    wrpT[(size_t)n * (3 * DIM) + k] = f2bf(Wrp[k * DIM + n]);
  }
}

// ---------------------------------------------------------------- K1: QKV GEMM
// grid (512, 9): z = branch*3 + {q,k,v}. 64-row tile, full K=256 in LDS (bf16).
#define LDA1 264
__global__ __launch_bounds__(256) void qkv_gemm(
    const float* __restrict__ f1, const float* __restrict__ f1p,
    const float* __restrict__ f2, const float* __restrict__ f2p,
    const float* __restrict__ f3, const float* __restrict__ f3p,
    const float* __restrict__ f4, const float* __restrict__ f4p,
    const float* __restrict__ bq, const float* __restrict__ bk,
    const float* __restrict__ bv, const unsigned short* __restrict__ wqkvT,
    float* __restrict__ qbuf, float* __restrict__ kbuf, float* __restrict__ vbuf) {
  __shared__ unsigned short sA[64 * LDA1];
  const int z = blockIdx.y;
  const int br = z / 3, part = z % 3;
  const int row0 = blockIdx.x * 64;
  const float *A0, *A1, *bias;
  float* outp;
  if (br == 0)      { A0 = f2; A1 = f2p; }
  else if (br == 1) { A0 = f3; A1 = f3p; }
  else              { A0 = f4; A1 = f4p; }
  if (part == 0)      { A0 = f1; A1 = f1p;  bias = bq + br * DIM; outp = qbuf; }
  else if (part == 1) {                     bias = bk + br * DIM; outp = kbuf; }
  else                { A1 = nullptr;       bias = bv + br * DIM; outp = vbuf; }
  outp += (size_t)br * MROWS * DIM;

  const int tid = threadIdx.x;
  #pragma unroll
  for (int i = 0; i < 16; ++i) {
    int lin = tid + i * 256;                 // float4 index, < 4096
    int r = lin >> 6, c4 = lin & 63;
    size_t g = (size_t)(row0 + r) * DIM + c4 * 4;
    f32x4 x = *(const f32x4*)(A0 + g);
    if (A1) { f32x4 y = *(const f32x4*)(A1 + g); x += y; }
    u16x4 h;
    h.x = f2bf(x.x); h.y = f2bf(x.y); h.z = f2bf(x.z); h.w = f2bf(x.w);
    *(u16x4*)&sA[r * LDA1 + c4 * 4] = h;
  }
  __syncthreads();

  const int wave = tid >> 5, lane = tid & 31;
  const int rb = wave >> 1, ch = wave & 1;   // 16-row block, 128-col half
  const unsigned short* wt = wqkvT + (size_t)z * DIM * DIM;
  f32x8 acc[8] = {};
  #pragma unroll
  for (int kc = 0; kc < 8; ++kc) {
    bf16x16 a = load_a_frag(&sA[(rb * 16) * LDA1 + kc * 32], LDA1);
    #pragma unroll
    for (int t = 0; t < 8; ++t) {
      bf16x16 b = load_b_frag(wt + (size_t)(ch * 128 + t * 16) * DIM + kc * 32, DIM);
      acc[t] = wmma_bf16(a, b, acc[t]);
    }
  }
  const int hf = lane >> 4, l15 = lane & 15;
  #pragma unroll
  for (int t = 0; t < 8; ++t) {
    int col = ch * 128 + t * 16 + l15;
    float bsv = bias[col];
    #pragma unroll
    for (int r = 0; r < 8; ++r) {
      int mrow = rb * 16 + r + 8 * hf;
      outp[(size_t)(row0 + mrow) * DIM + col] = acc[t][r] + bsv;
    }
  }
}

// ---------------------------------------------------------------- K2: q stats
__global__ __launch_bounds__(256) void q_stats_partial(
    const float* __restrict__ qbuf, float* __restrict__ pm, float* __restrict__ ps) {
  const int chunk = blockIdx.x, b = blockIdx.y, br = blockIdx.z;
  const int d = threadIdx.x;
  const float* p = qbuf + ((size_t)br * MROWS + (size_t)b * SEQ + chunk * 1024) * DIM + d;
  float mx = -3.4e38f, s = 0.f;
  #pragma unroll 4
  for (int i = 0; i < 1024; ++i) {
    float x = p[(size_t)i * DIM];
    float nm = fmaxf(mx, x);
    s = s * __expf(mx - nm) + __expf(x - nm);
    mx = nm;
  }
  size_t o = (((size_t)(br * BSZ + b)) * 8 + chunk) * DIM + d;
  pm[o] = mx; ps[o] = s;
}

__global__ __launch_bounds__(256) void q_stats_combine(
    const float* __restrict__ pm, const float* __restrict__ ps,
    float* __restrict__ mArr, float* __restrict__ sArr) {
  int idx = blockIdx.x * 256 + threadIdx.x;       // < 3072
  int bbr = idx >> 8, d = idx & 255;
  float mx = -3.4e38f;
  #pragma unroll
  for (int c = 0; c < 8; ++c) mx = fmaxf(mx, pm[((size_t)bbr * 8 + c) * DIM + d]);
  float s = 0.f;
  #pragma unroll
  for (int c = 0; c < 8; ++c)
    s += ps[((size_t)bbr * 8 + c) * DIM + d] * __expf(pm[((size_t)bbr * 8 + c) * DIM + d] - mx);
  mArr[idx] = mx; sArr[idx] = s;
}

// ------------------------------------------------- K2a: k softmax + kT/vT bf16
#define LDT 265
__global__ __launch_bounds__(256) void kv_transform(
    const float* __restrict__ kbuf, const float* __restrict__ vbuf,
    unsigned short* __restrict__ ksmT, unsigned short* __restrict__ vT) {
  __shared__ float tile[32 * LDT];
  const int n0 = blockIdx.x * 32, b = blockIdx.y, br = blockIdx.z;
  const int tid = threadIdx.x;
  const size_t inbase = ((size_t)br * MROWS + (size_t)b * SEQ + n0) * DIM;
  // ---- K: load, per-head softmax over 32 channels, transposed bf16 store
  #pragma unroll
  for (int i = 0; i < 8; ++i) {
    int lin = tid + i * 256;
    int r = lin >> 6, c4 = lin & 63;
    f32x4 x = *(const f32x4*)(kbuf + inbase + (size_t)r * DIM + c4 * 4);
    float* dst = &tile[r * LDT + c4 * 4];
    dst[0] = x.x; dst[1] = x.y; dst[2] = x.z; dst[3] = x.w;
  }
  __syncthreads();
  {
    const int h = tid >> 5, row = tid & 31;
    const float* src = &tile[row * LDT + h * HD];
    float mx = -3.4e38f;
    #pragma unroll
    for (int c = 0; c < HD; ++c) mx = fmaxf(mx, src[c]);
    float s = 0.f;
    #pragma unroll
    for (int c = 0; c < HD; ++c) s += __expf(src[c] - mx);
    float inv = 1.f / s;
    size_t obase = (((size_t)(br * BSZ + b) * HEADS + h) * HD) * SEQ + n0 + row;
    #pragma unroll
    for (int c = 0; c < HD; ++c)
      ksmT[obase + (size_t)c * SEQ] = f2bf(__expf(src[c] - mx) * inv);
  }
  __syncthreads();
  // ---- V: load, transposed bf16 store
  #pragma unroll
  for (int i = 0; i < 8; ++i) {
    int lin = tid + i * 256;
    int r = lin >> 6, c4 = lin & 63;
    f32x4 x = *(const f32x4*)(vbuf + inbase + (size_t)r * DIM + c4 * 4);
    float* dst = &tile[r * LDT + c4 * 4];
    dst[0] = x.x; dst[1] = x.y; dst[2] = x.z; dst[3] = x.w;
  }
  __syncthreads();
  {
    const int h = tid >> 5, row = tid & 31;
    const float* src = &tile[row * LDT + h * HD];
    size_t obase = (((size_t)(br * BSZ + b) * HEADS + h) * HD) * SEQ + n0 + row;
    #pragma unroll
    for (int d = 0; d < HD; ++d)
      vT[obase + (size_t)d * SEQ] = f2bf(src[d]);
  }
}

// -------------------------------------- K2c: ctx = ksm @ v^T (split-K WMMA), /S
__global__ __launch_bounds__(256) void ctx_gemm(
    const unsigned short* __restrict__ ksmT, const unsigned short* __restrict__ vT,
    const float* __restrict__ sArr, unsigned short* __restrict__ ctxT) {
  __shared__ float red[8 * HD * HD];
  const int h = blockIdx.x, b = blockIdx.y, br = blockIdx.z;
  const int tid = threadIdx.x, wave = tid >> 5, lane = tid & 31;
  const size_t bh = (size_t)(br * BSZ + b) * HEADS + h;
  const unsigned short* ks = ksmT + bh * HD * SEQ;
  const unsigned short* vt = vT   + bh * HD * SEQ;
  f32x8 a00 = {}, a01 = {}, a10 = {}, a11 = {};
  const int off0 = wave * 1024;                     // split-K: 8 waves x 1024
  #pragma unroll 2
  for (int kc = 0; kc < 32; ++kc) {
    int off = off0 + kc * 32;
    bf16x16 A0 = load_a_frag(ks + off, SEQ);
    bf16x16 A1 = load_a_frag(ks + (size_t)16 * SEQ + off, SEQ);
    bf16x16 B0 = load_b_frag(vt + off, SEQ);
    bf16x16 B1 = load_b_frag(vt + (size_t)16 * SEQ + off, SEQ);
    a00 = wmma_bf16(A0, B0, a00); a01 = wmma_bf16(A0, B1, a01);
    a10 = wmma_bf16(A1, B0, a10); a11 = wmma_bf16(A1, B1, a11);
  }
  const int hf = lane >> 4, l15 = lane & 15;
  #pragma unroll
  for (int r = 0; r < 8; ++r) {
    int c0 = r + 8 * hf;
    red[wave * 1024 + c0 * HD + l15]             = a00[r];
    red[wave * 1024 + c0 * HD + 16 + l15]        = a01[r];
    red[wave * 1024 + (16 + c0) * HD + l15]      = a10[r];
    red[wave * 1024 + (16 + c0) * HD + 16 + l15] = a11[r];
  }
  __syncthreads();
  #pragma unroll
  for (int i = 0; i < 4; ++i) {
    int e = tid + i * 256;                          // < 1024
    int c = e >> 5, d = e & 31;
    float s = 0.f;
    #pragma unroll
    for (int w = 0; w < 8; ++w) s += red[w * 1024 + c * HD + d];
    float inv = 1.f / sArr[(size_t)(br * BSZ + b) * DIM + h * HD + c];
    ctxT[bh * HD * HD + (size_t)d * HD + c] = f2bf(s * inv);  // store ctx' transposed
  }
}

// ------------------- K3: fused exp(q-m) -> per-head WMMA -> 768x256 WMMA + bias
#define LDE 776
__global__ __launch_bounds__(256) void att_out(
    const float* __restrict__ qbuf, const float* __restrict__ mArr,
    const unsigned short* __restrict__ ctxT, const unsigned short* __restrict__ wrpT,
    const float* __restrict__ brp, float* __restrict__ out) {
  __shared__ unsigned short E[16 * LDE];
  __shared__ unsigned short Af[16 * LDE];
  const int row0 = blockIdx.x * 16;
  const int b = row0 / SEQ;
  const int tid = threadIdx.x, wave = tid >> 5, lane = tid & 31;
  const int hf = lane >> 4, l15 = lane & 15;
  // stage 1: E[r][br*256+c] = exp(q - m) in bf16
  #pragma unroll
  for (int i = 0; i < 12; ++i) {
    int lin = tid + i * 256;                        // < 3072 float4s
    int br = lin >> 10;
    int rem = lin & 1023;
    int r = rem >> 6, c4 = rem & 63;
    f32x4 x  = *(const f32x4*)(qbuf + ((size_t)br * MROWS + row0 + r) * DIM + c4 * 4);
    f32x4 mm = *(const f32x4*)(mArr + (size_t)(br * BSZ + b) * DIM + c4 * 4);
    u16x4 h;
    h.x = f2bf(__expf(x.x - mm.x)); h.y = f2bf(__expf(x.y - mm.y));
    h.z = f2bf(__expf(x.z - mm.z)); h.w = f2bf(__expf(x.w - mm.w));
    *(u16x4*)&E[r * LDE + br * 256 + c4 * 4] = h;
  }
  __syncthreads();
  // stage 2: attention apply per (branch, head): Af = E_head(16x32) x ctx'(32x32)
  #pragma unroll
  for (int j = 0; j < 3; ++j) {
    int idx = wave * 3 + j;                         // 0..23
    int br = idx >> 3, h = idx & 7;
    int colb = br * 256 + h * HD;
    bf16x16 a = load_a_frag(&E[colb], LDE);
    const unsigned short* ct = ctxT + ((size_t)(br * BSZ + b) * HEADS + h) * HD * HD;
    bf16x16 b0 = load_b_frag(ct, HD);
    bf16x16 b1 = load_b_frag(ct + (size_t)16 * HD, HD);
    f32x8 c0 = {}, c1 = {};
    c0 = wmma_bf16(a, b0, c0);
    c1 = wmma_bf16(a, b1, c1);
    #pragma unroll
    for (int r = 0; r < 8; ++r) {
      int mrow = r + 8 * hf;
      Af[mrow * LDE + colb + l15]      = f2bf(c0[r]);
      Af[mrow * LDE + colb + 16 + l15] = f2bf(c1[r]);
    }
  }
  __syncthreads();
  // stage 3: out(16x256) = Af(16x768) x Wrp(768x256) + brp; wave -> 32 columns
  f32x8 acc0 = {}, acc1 = {};
  const unsigned short* w0 = wrpT + (size_t)(wave * 32) * 768;
  const unsigned short* w1 = wrpT + (size_t)(wave * 32 + 16) * 768;
  #pragma unroll
  for (int kc = 0; kc < 24; ++kc) {
    bf16x16 a = load_a_frag(&Af[kc * 32], LDE);
    acc0 = wmma_bf16(a, load_b_frag(w0 + kc * 32, 768), acc0);
    acc1 = wmma_bf16(a, load_b_frag(w1 + kc * 32, 768), acc1);
  }
  int col0 = wave * 32 + l15;
  float bias0 = brp[col0], bias1 = brp[col0 + 16];
  #pragma unroll
  for (int r = 0; r < 8; ++r) {
    int mrow = r + 8 * hf;
    out[(size_t)(row0 + mrow) * DIM + col0]      = acc0[r] + bias0;
    out[(size_t)(row0 + mrow) * DIM + col0 + 16] = acc1[r] + bias1;
  }
}

// ---------------------------------------------------------------- launch
extern "C" void kernel_launch(void* const* d_in, const int* in_sizes, int n_in,
                              void* d_out, int out_size, void* d_ws, size_t ws_size,
                              hipStream_t stream) {
  (void)in_sizes; (void)n_in; (void)out_size; (void)ws_size;
  const float* f1  = (const float*)d_in[0];
  const float* f2  = (const float*)d_in[1];
  const float* f3  = (const float*)d_in[2];
  const float* f4  = (const float*)d_in[3];
  const float* f1p = (const float*)d_in[4];
  const float* f2p = (const float*)d_in[5];
  const float* f3p = (const float*)d_in[6];
  const float* f4p = (const float*)d_in[7];
  const float* Wq  = (const float*)d_in[8];
  const float* bq  = (const float*)d_in[9];
  const float* Wk  = (const float*)d_in[10];
  const float* bk  = (const float*)d_in[11];
  const float* Wv  = (const float*)d_in[12];
  const float* bv  = (const float*)d_in[13];
  const float* Wrp = (const float*)d_in[14];
  const float* brp = (const float*)d_in[15];
  float* outp = (float*)d_out;

  char* ws = (char*)d_ws;
  const size_t SZQ = (size_t)3 * MROWS * DIM * sizeof(float);        // 100663296
  const size_t SZT = (size_t)3 * BSZ * HEADS * HD * SEQ * 2;         // 50331648
  float* qbuf = (float*)(ws);
  float* kbuf = (float*)(ws + SZQ);
  float* vbuf = (float*)(ws + 2 * SZQ);
  unsigned short* ksmT  = (unsigned short*)(ws + 3 * SZQ);
  unsigned short* vT    = (unsigned short*)(ws + 3 * SZQ + SZT);
  unsigned short* ctxT  = (unsigned short*)(ws + 3 * SZQ + 2 * SZT);
  unsigned short* wqkvT = (unsigned short*)(ws + 3 * SZQ + 2 * SZT + 196608);
  unsigned short* wrpT  = (unsigned short*)(ws + 3 * SZQ + 2 * SZT + 196608 + 1179648);
  float* mArr = (float*)(ws + 3 * SZQ + 2 * SZT + 196608 + 1179648 + 393216);
  float* sArr = mArr + 3 * BSZ * DIM;
  float* pm   = sArr + 3 * BSZ * DIM;
  float* ps   = pm + (size_t)3 * BSZ * 8 * DIM;

  prep_weights<<<3072, 256, 0, stream>>>(Wq, Wk, Wv, Wrp, wqkvT, wrpT);
  qkv_gemm<<<dim3(MROWS / 64, 9), 256, 0, stream>>>(
      f1, f1p, f2, f2p, f3, f3p, f4, f4p, bq, bk, bv, wqkvT, qbuf, kbuf, vbuf);
  q_stats_partial<<<dim3(8, BSZ, 3), 256, 0, stream>>>(qbuf, pm, ps);
  q_stats_combine<<<12, 256, 0, stream>>>(pm, ps, mArr, sArr);
  kv_transform<<<dim3(SEQ / 32, BSZ, 3), 256, 0, stream>>>(kbuf, vbuf, ksmT, vT);
  ctx_gemm<<<dim3(HEADS, BSZ, 3), 256, 0, stream>>>(ksmT, vT, sArr, ctxT);
  att_out<<<MROWS / 16, 256, 0, stream>>>(qbuf, mArr, ctxT, wrpT, brp, outp);
}
